// Attention_Rel_Vec_15272903705327
// MI455X (gfx1250) — matrix operand order repaired and checked by reading, not compiled, not run on hardware
//
#include <hip/hip_runtime.h>

typedef __attribute__((ext_vector_type(16))) _Float16 v16h;
typedef __attribute__((ext_vector_type(8)))  _Float16 v8h;
typedef __attribute__((ext_vector_type(8)))  float    v8f;

#define BATCH 4
#define SEQ   2048
#define EMB   512
#define NH    8
#define HDIM  64

__device__ __forceinline__ v16h pack16(v8h lo, v8h hi) {
  v16h r;
#pragma unroll
  for (int t = 0; t < 8; ++t) { r[t] = lo[t]; r[t + 8] = hi[t]; }
  return r;
}

__device__ __forceinline__ v8f wmma_f16(v16h a, v16h b, v8f c) {
  // D = A(16x32 f16) * B(32x16 f16) + C(16x16 f32)
  return __builtin_amdgcn_wmma_f32_16x16x32_f16(false, a, false, b, (short)0, c,
                                                false, false);
}

// ---------------------------------------------------------------------------
// fp32 -> fp16 conversion
// ---------------------------------------------------------------------------
__global__ void cvt_f32_f16(const float* __restrict__ src,
                            _Float16* __restrict__ dst, int n) {
  int i = blockIdx.x * blockDim.x + threadIdx.x;
  if (i < n) dst[i] = (_Float16)src[i];
}

// ---------------------------------------------------------------------------
// out = x @ W^T  (x:[8192,512] f16, W:[512,512] f16)
// One 32x64 output tile per wave: 2 A-frags x 4 B-frags, 8 WMMA per K-step.
// vt==0: scatter into [B,H,S,HD]; vt==1: scatter transposed into [B,H,HD,S].
// ---------------------------------------------------------------------------
__global__ void __launch_bounds__(256)
qkv_gemm(const _Float16* __restrict__ X, const _Float16* __restrict__ W,
         _Float16* __restrict__ Dst, int vt) {
  const int gw   = (blockIdx.x * blockDim.x + threadIdx.x) >> 5;  // 0..2047
  const int lane = threadIdx.x & 31;
  const int lm   = lane & 15;
  const int hl8  = (lane < 16) ? 0 : 8;   // accumulator half-lane row offset
  const int hsel = (lane < 16) ? 0 : 16;  // B-frag half-lane K offset

  const int tn = gw >> 3;                 // row tile (32 rows) 0..255
  const int to = gw & 7;                  // col tile (64 cols) 0..7
  const int n0 = tn * 32, o0 = to * 64;

  const _Float16* arow0 = X + (size_t)(n0 + lm) * EMB;
  const _Float16* arow1 = X + (size_t)(n0 + 16 + lm) * EMB;
  const _Float16* brow  = W + (size_t)(o0 + lm) * EMB;

  v8f acc[2][4] = {};
  for (int k0 = 0; k0 < EMB; k0 += 32) {
    __builtin_prefetch(arow0 + k0 + 256, 0, 1);
    __builtin_prefetch(arow1 + k0 + 256, 0, 1);
    v16h a0 = pack16(*(const v8h*)(arow0 + k0 + hsel),
                     *(const v8h*)(arow0 + k0 + hsel + 8));
    v16h a1 = pack16(*(const v8h*)(arow1 + k0 + hsel),
                     *(const v8h*)(arow1 + k0 + hsel + 8));
#pragma unroll
    for (int c = 0; c < 4; ++c) {
      v16h bf = *(const v16h*)(brow + (size_t)(c * 16) * EMB + k0 + hsel);
      acc[0][c] = wmma_f16(a0, bf, acc[0][c]);
      acc[1][c] = wmma_f16(a1, bf, acc[1][c]);
    }
  }

#pragma unroll
  for (int c = 0; c < 4; ++c) {
    const int o = o0 + c * 16 + lm;
    const int h = o >> 6, hd = o & (HDIM - 1);
#pragma unroll
    for (int g = 0; g < 2; ++g)
#pragma unroll
      for (int r = 0; r < 8; ++r) {
        const int n  = n0 + g * 16 + r + hl8;
        const int bb = n >> 11, s = n & (SEQ - 1);
        const size_t di = vt
            ? (((size_t)bb * NH + h) * HDIM + hd) * SEQ + s
            : (((size_t)bb * NH + h) * SEQ + s) * HDIM + hd;
        Dst[di] = (_Float16)acc[g][c][r];
      }
  }
}

// ---------------------------------------------------------------------------
// Fused relative-position attention (flash style).
// One wave owns 16 query rows of one (b,h); streams keys in chunks of 32.
// V is pre-transposed [B,H,HD,S] so P@V B-fragments are contiguous loads.
// ---------------------------------------------------------------------------
__global__ void __launch_bounds__(256)
attn_kernel(const _Float16* __restrict__ Qh, const _Float16* __restrict__ Kh,
            const _Float16* __restrict__ Vt, const _Float16* __restrict__ Erh,
            float* __restrict__ Out) {
  __shared__ alignas(64) float    gwin_s[8][16][48];   // 24 KB
  __shared__ alignas(64) _Float16 p_s[8][16][32];      //  8 KB

  const int wv   = threadIdx.x >> 5;
  const int lane = threadIdx.x & 31;
  const int gw   = blockIdx.x * 8 + wv;        // 0..4095
  const int qt   = gw & 127;
  const int h    = (gw >> 7) & (NH - 1);
  const int b    = gw >> 10;
  const int s0   = qt * 16;
  const int lm   = lane & 15;
  const int hl8  = (lane < 16) ? 0 : 8;
  const int hsel = (lane < 16) ? 0 : 16;

  const _Float16* Qb  = Qh + (((size_t)b * NH + h) * SEQ) * HDIM;
  const _Float16* Kb  = Kh + (((size_t)b * NH + h) * SEQ) * HDIM;
  const _Float16* Vtb = Vt + (((size_t)b * NH + h) * HDIM) * SEQ;

  // Q A-fragments: 16x64 split into two K=32 steps
  v16h qa[2];
  {
    const _Float16* qrow = Qb + (size_t)(s0 + lm) * HDIM;
#pragma unroll
    for (int kk = 0; kk < 2; ++kk) {
      v8h lo = *(const v8h*)(qrow + kk * 32 + hsel);
      v8h hi = *(const v8h*)(qrow + kk * 32 + hsel + 8);
      qa[kk] = pack16(lo, hi);
    }
  }

  float m_i[8], l_i[8];
  v8f   o_acc[4] = {};
#pragma unroll
  for (int r = 0; r < 8; ++r) { m_i[r] = -3.0e38f; l_i[r] = 0.0f; }

  const float scale = 0.04419417382415922f;  // 512^-0.5
  v8f gcarry = {};                           // rel-window subtile overlap

  for (int t0 = 0; t0 < SEQ; t0 += 32) {
    // ---- S = Q @ K^T : 16x32 tile (2 N-subtiles x 2 K-steps) ----
    v8f sacc[2] = {};
#pragma unroll
    for (int sub = 0; sub < 2; ++sub) {
      const _Float16* krow = Kb + (size_t)(t0 + sub * 16 + lm) * HDIM;
#pragma unroll
      for (int kk = 0; kk < 2; ++kk) {
        v16h bf = *(const v16h*)(krow + kk * 32 + hsel);
        sacc[sub] = wmma_f16(qa[kk], bf, sacc[sub]);
      }
    }

    // ---- relative window G = Q @ Er_win^T : 16x48 ----
    // subtile 0 of this chunk == subtile 2 of the previous chunk (carry).
    const int lbase = (SEQ - 1) + (t0 - s0) - 15;
#pragma unroll
    for (int sub = 0; sub < 3; ++sub) {
      v8f g;
      if (sub == 0 && t0 != 0) {
        g = gcarry;
      } else {
        int row = lbase + sub * 16 + lm;
        row = row < 0 ? 0 : (row > SEQ - 1 ? SEQ - 1 : row);
        const _Float16* erow = Erh + (size_t)row * HDIM;
        v8f gz = {};
#pragma unroll
        for (int kk = 0; kk < 2; ++kk) {
          v16h bf = *(const v16h*)(erow + kk * 32 + hsel);
          gz = wmma_f16(qa[kk], bf, gz);
        }
        g = gz;
      }
      if (sub == 2) gcarry = g;
#pragma unroll
      for (int r = 0; r < 8; ++r)
        gwin_s[wv][r + hl8][sub * 16 + lm] = g[r];
    }
    __syncthreads();

    // ---- scores + online softmax (rows live in 16-lane groups) ----
#pragma unroll
    for (int r = 0; r < 8; ++r) {
      const int i  = r + hl8;
      const int j0 = lm, j1 = lm + 16;
      float rel0 = (t0 + j0 <= s0 + i) ? gwin_s[wv][i][j0 - i + 15] : 0.0f;
      float rel1 = (t0 + j1 <= s0 + i) ? gwin_s[wv][i][j1 - i + 15] : 0.0f;
      float v0 = (sacc[0][r] + rel0) * scale;
      float v1 = (sacc[1][r] + rel1) * scale;
      float mx = fmaxf(v0, v1);
#pragma unroll
      for (int msk = 1; msk <= 8; msk <<= 1)
        mx = fmaxf(mx, __shfl_xor(mx, msk, 32));
      const float mnew = fmaxf(m_i[r], mx);
      const float p0 = __expf(v0 - mnew);
      const float p1 = __expf(v1 - mnew);
      float rs = p0 + p1;
#pragma unroll
      for (int msk = 1; msk <= 8; msk <<= 1)
        rs += __shfl_xor(rs, msk, 32);
      const float alpha = __expf(m_i[r] - mnew);
      l_i[r] = l_i[r] * alpha + rs;
      m_i[r] = mnew;
#pragma unroll
      for (int f = 0; f < 4; ++f) o_acc[f][r] *= alpha;
      p_s[wv][i][j0] = (_Float16)p0;
      p_s[wv][i][j1] = (_Float16)p1;
    }
    __syncthreads();

    // ---- reload P in A-fragment layout (16x32, K = 32 keys) ----
    v16h pa;
    {
      const _Float16* prow = &p_s[wv][lm][0];
      v8h lo = *(const v8h*)(prow + hsel);
      v8h hi = *(const v8h*)(prow + hsel + 8);
      pa = pack16(lo, hi);
    }

    // ---- O += P @ V : 4 N-subtiles over HD=64, contiguous B loads ----
#pragma unroll
    for (int f = 0; f < 4; ++f) {
      const _Float16* vrow = Vtb + (size_t)(f * 16 + lm) * SEQ + t0 + hsel;
      v16h bf = *(const v16h*)vrow;
      o_acc[f] = wmma_f16(pa, bf, o_acc[f]);
    }
    __syncthreads();
  }

  // ---- epilogue: normalize and write [B,S,E] f32 ----
  float rcp[8];
#pragma unroll
  for (int r = 0; r < 8; ++r) rcp[r] = 1.0f / l_i[r];
#pragma unroll
  for (int f = 0; f < 4; ++f) {
    const int hd = f * 16 + lm;
#pragma unroll
    for (int r = 0; r < 8; ++r) {
      const int i = r + hl8;
      Out[((size_t)b * SEQ + s0 + i) * EMB + h * HDIM + hd] =
          o_acc[f][r] * rcp[r];
    }
  }
}

// ---------------------------------------------------------------------------
// LayerNorm over E=512, one wave per row
// ---------------------------------------------------------------------------
__global__ void __launch_bounds__(256)
ln_kernel(const float* __restrict__ X, const float* __restrict__ gamma,
          const float* __restrict__ beta, float* __restrict__ Out) {
  const int row  = (blockIdx.x * blockDim.x + threadIdx.x) >> 5;
  const int lane = threadIdx.x & 31;
  const float* xr = X + (size_t)row * EMB;
  float v[16];
  float sum = 0.0f;
#pragma unroll
  for (int t = 0; t < 16; ++t) { v[t] = xr[lane + t * 32]; sum += v[t]; }
#pragma unroll
  for (int msk = 1; msk <= 16; msk <<= 1) sum += __shfl_xor(sum, msk, 32);
  const float mu = sum * (1.0f / EMB);
  float sq = 0.0f;
#pragma unroll
  for (int t = 0; t < 16; ++t) { float d = v[t] - mu; sq += d * d; }
#pragma unroll
  for (int msk = 1; msk <= 16; msk <<= 1) sq += __shfl_xor(sq, msk, 32);
  const float rstd = rsqrtf(sq * (1.0f / EMB) + 1e-5f);
  float* outr = Out + (size_t)row * EMB;
#pragma unroll
  for (int t = 0; t < 16; ++t) {
    const int c = lane + t * 32;
    outr[c] = (v[t] - mu) * rstd * gamma[c] + beta[c];
  }
}

// ---------------------------------------------------------------------------
extern "C" void kernel_launch(void* const* d_in, const int* in_sizes, int n_in,
                              void* d_out, int out_size, void* d_ws,
                              size_t ws_size, hipStream_t stream) {
  (void)in_sizes; (void)n_in; (void)out_size; (void)ws_size;
  const float* x     = (const float*)d_in[0];
  const float* Wq    = (const float*)d_in[1];
  const float* Wk    = (const float*)d_in[2];
  const float* Wv    = (const float*)d_in[3];
  const float* Er    = (const float*)d_in[4];
  const float* gamma = (const float*)d_in[5];
  const float* beta  = (const float*)d_in[6];

  const size_t nTok = (size_t)BATCH * SEQ;          // 8192
  char* ws = (char*)d_ws;
  _Float16* xh  = (_Float16*)ws;  ws += nTok * EMB * sizeof(_Float16);
  _Float16* wqh = (_Float16*)ws;  ws += (size_t)EMB * EMB * sizeof(_Float16);
  _Float16* wkh = (_Float16*)ws;  ws += (size_t)EMB * EMB * sizeof(_Float16);
  _Float16* wvh = (_Float16*)ws;  ws += (size_t)EMB * EMB * sizeof(_Float16);
  _Float16* erh = (_Float16*)ws;  ws += (size_t)SEQ * HDIM * sizeof(_Float16);
  _Float16* qh  = (_Float16*)ws;  ws += nTok * EMB * sizeof(_Float16);
  _Float16* kh  = (_Float16*)ws;  ws += nTok * EMB * sizeof(_Float16);
  _Float16* vth = (_Float16*)ws;  ws += nTok * EMB * sizeof(_Float16);
  float*    ao  = (float*)ws;     ws += nTok * EMB * sizeof(float);

  const int nx = (int)(nTok * EMB);       // 4,194,304
  const int nw = EMB * EMB;               // 262,144
  const int ne = SEQ * HDIM;              // 131,072
  cvt_f32_f16<<<(nx + 255) / 256, 256, 0, stream>>>(x,  xh,  nx);
  cvt_f32_f16<<<(nw + 255) / 256, 256, 0, stream>>>(Wq, wqh, nw);
  cvt_f32_f16<<<(nw + 255) / 256, 256, 0, stream>>>(Wk, wkh, nw);
  cvt_f32_f16<<<(nw + 255) / 256, 256, 0, stream>>>(Wv, wvh, nw);
  cvt_f32_f16<<<(ne + 255) / 256, 256, 0, stream>>>(Er, erh, ne);

  // 2048 waves per matrix (32x64 tiles) / 8 waves per block = 256 blocks
  qkv_gemm<<<256, 256, 0, stream>>>(xh, wqh, qh, 0);
  qkv_gemm<<<256, 256, 0, stream>>>(xh, wkh, kh, 0);
  qkv_gemm<<<256, 256, 0, stream>>>(xh, wvh, vth, 1);  // V transposed

  // 4*8*128 = 4096 query tiles / 8 waves per block = 512 blocks
  attn_kernel<<<512, 256, 0, stream>>>(qh, kh, vth, erh, ao);

  // 8192 rows / 8 waves per block = 1024 blocks
  ln_kernel<<<1024, 256, 0, stream>>>(ao, gamma, beta, (float*)d_out);
}
